// MHA_80736795230299
// MI455X (gfx1250) — matrix operand order repaired and checked by reading, not compiled, not run on hardware
//
#include <hip/hip_runtime.h>
#include <hip/hip_bf16.h>
#include <math.h>

// ---------------------------------------------------------------------------
// MHA forward for MI455X (gfx1250, wave32, WMMA 16x16x32 f16 -> f32 accum)
//   N=2, S=2048, d_model=1024, H=16, D=64
//   d_out = [hidden (2*2048*1024 f32)] ++ [attn (2*16*2048*2048 f32)]
//   d_ws  = Qf|Kf|Vf|CTX (f32, 16MB each) ++ Qh|Kh|Vh (f16 head-major, 8MB each)
//
// Fused attention kernel keeps the 16x2048 logit strip in LDS (320KB/WGP):
// attn is written exactly once (536MB) instead of 3.5 passes -> ~45us HBM floor.
// K tiles are pulled into LDS by the Tensor Data Mover (tensor_load_to_lds).
// ---------------------------------------------------------------------------

#define NB   2
#define SS   2048
#define DM   1024
#define NH   16
#define DH   64
#define MT   (NB * SS)          // 4096 rows for the projection GEMMs
#define SLS  2064               // logit strip LDS row stride (f32), 2048+16 pad

typedef __attribute__((ext_vector_type(16))) _Float16     v16h;
typedef __attribute__((ext_vector_type(2)))  _Float16     v2h;
typedef __attribute__((ext_vector_type(8)))  float        v8f;
typedef __attribute__((ext_vector_type(4)))  unsigned int v4u;
typedef __attribute__((ext_vector_type(8)))  int          v8i;
typedef __attribute__((ext_vector_type(4)))  int          v4i;

// ---- WMMA fragment gathers (layouts per CDNA5 ISA 7.12.2, wave32) ---------
// A (16-bit, 16xK): m = lane&15, g = lane>>4; element i: K = i+8g+(i>=8?8:0)
__device__ __forceinline__ v16h load_a_frag(const _Float16* lds, int stride,
                                            int m, int g, int dbase) {
    v16h a;
    const _Float16* rowp = lds + m * stride + dbase;
#pragma unroll
    for (int r = 0; r < 8; ++r) {
        int k0 = (r < 4) ? (8 * g + 2 * r) : (16 + 8 * g + 2 * (r - 4));
        a[2 * r]     = rowp[k0];
        a[2 * r + 1] = rowp[k0 + 1];
    }
    return a;
}

// B (16-bit, Kx16) staged as [N][K] in LDS: n = lane&15; element i: K = i+8g+(i>=8?8:0)
__device__ __forceinline__ v16h load_b_frag(const _Float16* lds, int stride,
                                            int n, int g, int dbase) {
    v16h b;
    const _Float16* rowp = lds + n * stride + dbase;
#pragma unroll
    for (int r = 0; r < 8; ++r) {
        int k0 = (r < 4) ? (8 * g + 2 * r) : (16 + 8 * g + 2 * (r - 4));
        b[2 * r]     = rowp[k0];
        b[2 * r + 1] = rowp[k0 + 1];
    }
    return b;
}

// A fragment built from f32 probabilities in LDS (convert on the fly)
__device__ __forceinline__ v16h a_frag_from_probs(const float* lds, int stride,
                                                  int m, int g, int kbase) {
    v16h a;
    const float* rowp = lds + m * stride + kbase;
#pragma unroll
    for (int r = 0; r < 8; ++r) {
        int k0 = (r < 4) ? (8 * g + 2 * r) : (16 + 8 * g + 2 * (r - 4));
        a[2 * r]     = (_Float16)rowp[k0];
        a[2 * r + 1] = (_Float16)rowp[k0 + 1];
    }
    return a;
}

// ---- Tensor Data Mover: 2-D f16 tile (64 wide x rows) -> LDS --------------
// D# per ISA 8.3/8.4: count=1, type=2, data_size=2B, tensor row = 64 elems,
// pad_interval=32 DWORDs (one 128B row), pad_amount=4 DWORDs (16B)
//  -> LDS row stride = 72 halves, matching load_b_frag(stride=72).
// This toolchain's builtin takes 6 args (clang-23 / therock-10.0 form):
//   (uint32x4 g0, int32x8 g1, int32x4 g2, int32x4 g3, int32x8 extra, i32 cpol)
__device__ __forceinline__ void tdm_load_k_tile(unsigned lds_off,
                                                const _Float16* gptr,
                                                int rows) {
    unsigned long long ga = (unsigned long long)(size_t)(const void*)gptr;
    v4u g0;
    g0[0] = 1u;                                   // count=1, user mode
    g0[1] = lds_off;                              // LDS byte address
    g0[2] = (unsigned)ga;                         // global_addr[31:0]
    g0[3] = (unsigned)((ga >> 32) & 0x01FFFFFFu)  // global_addr[56:32]
          | (2u << 30);                           // type=2 ("image")
    v8i g1;
    g1[0] = (int)((1u << 16)                      // data_size = 2 bytes
                | (1u << 20)                      // pad_enable
                | (4u << 22)                      // pad_interval = 32 DWORDs
                | (3u << 25));                    // pad_amount  = 4 DWORDs
    g1[1] = (int)(64u << 16);                     // tensor_dim0 = 64 (lo16)
    g1[2] = (int)((2048u & 0xFFFFu) << 16);       // dim0 hi | tensor_dim1 lo16
    g1[3] = (int)((2048u >> 16) | (64u << 16));   // dim1 hi | tile_dim0 = 64
    g1[4] = (int)(rows & 0xFFFF);                 // tile_dim1 | tile_dim2 = 0
    g1[5] = 64;                                   // tensor_dim0_stride = 64
    g1[6] = 0;
    g1[7] = 0;
    v4i z4 = {};
    v8i z8 = {};
    __builtin_amdgcn_tensor_load_to_lds(g0, g1, z4, z4, z8, 0);
}

// ---------------------------------------------------------------------------
// Kernel 1: generic GEMM  C[M,Nc] = A[M,K] @ B[K,Nc] (+ bias), f32 in/out,
// f16 WMMA inner product. 256 thr = 8 waves, 64x64 tile, 2 accum tiles/wave.
// ---------------------------------------------------------------------------
__global__ __launch_bounds__(256) void gemm_bias_kernel(
    const float* __restrict__ A, const float* __restrict__ B,
    const float* __restrict__ bias, float* __restrict__ C,
    int M, int Nc, int K)
{
    __shared__ _Float16 lA[64 * 40];   // [row 64][k 32] pad->40
    __shared__ _Float16 lBt[64 * 40];  // [col 64][k 32] pad->40 (transposed)

    const int tid  = threadIdx.x;
    const int wave = tid >> 5, lane = tid & 31;
    const int g    = lane >> 4, ln = lane & 15;
    const int mbase = blockIdx.y * 64, nbase = blockIdx.x * 64;
    const int tr = wave >> 1, tc0 = (wave & 1) * 2;

    v8f acc0 = {}; v8f acc1 = {};

    for (int kt = 0; kt < K; kt += 32) {
        if (kt + 32 < K) {
            __builtin_prefetch(A + (size_t)(mbase + (tid >> 2)) * K + kt + 32, 0, 0);
        }
#pragma unroll
        for (int e = 0; e < 2; ++e) {                     // A 64x32
            int l4 = tid * 2 + e;
            int row = l4 >> 3, c4 = (l4 & 7) * 4;
            const float4 v = *reinterpret_cast<const float4*>(
                A + (size_t)(mbase + row) * K + kt + c4);
            _Float16* d = lA + row * 40 + c4;
            d[0] = (_Float16)v.x; d[1] = (_Float16)v.y;
            d[2] = (_Float16)v.z; d[3] = (_Float16)v.w;
        }
#pragma unroll
        for (int e = 0; e < 2; ++e) {                     // B 32x64 transposed
            int l4 = tid * 2 + e;
            int krow = l4 >> 4, c4 = (l4 & 15) * 4;
            const float4 v = *reinterpret_cast<const float4*>(
                B + (size_t)(kt + krow) * Nc + nbase + c4);
            lBt[(c4 + 0) * 40 + krow] = (_Float16)v.x;
            lBt[(c4 + 1) * 40 + krow] = (_Float16)v.y;
            lBt[(c4 + 2) * 40 + krow] = (_Float16)v.z;
            lBt[(c4 + 3) * 40 + krow] = (_Float16)v.w;
        }
        __syncthreads();

        v16h a  = load_a_frag(lA, 40, tr * 16 + ln, g, 0);
        v16h b0 = load_b_frag(lBt, 40, tc0 * 16 + ln, g, 0);
        v16h b1 = load_b_frag(lBt, 40, (tc0 + 1) * 16 + ln, g, 0);
        acc0 = __builtin_amdgcn_wmma_f32_16x16x32_f16(false, a, false, b0,
                                                      (short)0, acc0, false, false);
        acc1 = __builtin_amdgcn_wmma_f32_16x16x32_f16(false, a, false, b1,
                                                      (short)0, acc1, false, false);
        __syncthreads();
    }

    const int row0 = mbase + tr * 16 + 8 * g;
    const int col0 = nbase + tc0 * 16 + ln;
    const float bv0 = bias ? bias[col0] : 0.0f;
    const float bv1 = bias ? bias[col0 + 16] : 0.0f;
#pragma unroll
    for (int r = 0; r < 8; ++r) {
        C[(size_t)(row0 + r) * Nc + col0]      = acc0[r] + bv0;
        C[(size_t)(row0 + r) * Nc + col0 + 16] = acc1[r] + bv1;
    }
}

// ---------------------------------------------------------------------------
// Kernel 2: RoPE + f16 pack + transpose to head-major (N,H,S,D).
// ---------------------------------------------------------------------------
__global__ __launch_bounds__(256) void rope_pack_kernel(
    const float* __restrict__ src, _Float16* __restrict__ dst,
    const float* __restrict__ cosb, const float* __restrict__ sinb)
{
    int i = blockIdx.x * blockDim.x + threadIdx.x;   // N*S*H*32 pairs
    int p = i & 31;
    int h = (i >> 5) & 15;
    int s = (i >> 9) & 2047;
    int n = i >> 20;
    float c  = cosb[s * 32 + p];
    float sn = sinb[s * 32 + p];
    size_t sb = ((size_t)(n * SS + s)) * DM + h * DH + 2 * p;
    float xr = src[sb], xi = src[sb + 1];
    size_t ob = (((size_t)(n * NH + h)) * SS + s) * DH + 2 * p;
    dst[ob]     = (_Float16)(xr * c - xi * sn);
    dst[ob + 1] = (_Float16)(xr * sn + xi * c);
}

// V: f16 pack + transpose to head-major (no rotation)
__global__ __launch_bounds__(256) void pack_v_kernel(
    const float* __restrict__ src, _Float16* __restrict__ dst)
{
    int i = blockIdx.x * blockDim.x + threadIdx.x;
    int p = i & 31;
    int h = (i >> 5) & 15;
    int s = (i >> 9) & 2047;
    int n = i >> 20;
    size_t sb = ((size_t)(n * SS + s)) * DM + h * DH + 2 * p;
    size_t ob = (((size_t)(n * NH + h)) * SS + s) * DH + 2 * p;
    dst[ob]     = (_Float16)src[sb];
    dst[ob + 1] = (_Float16)src[sb + 1];
}

// ---------------------------------------------------------------------------
// Kernel 3: fused attention. One block per (n*H+h, 16-query strip).
//   phase 1: scores via WMMA (K tiles DMA'd by TDM), logits in LDS
//   phase 2: softmax over the LDS strip
//   phase 3: write attn probabilities to d_out exactly once (zeros past nk)
//   phase 4: ctx = P @ V via WMMA, split-K over 8 waves, LDS reduction
// ---------------------------------------------------------------------------
__global__ __launch_bounds__(256) void attn_fused_kernel(
    const _Float16* __restrict__ qh, const _Float16* __restrict__ kh,
    const _Float16* __restrict__ vh, float* __restrict__ attn,
    float* __restrict__ ctx)
{
    extern __shared__ char smem[];
    float*    sLog = (float*)smem;                               // 16*2064*4 = 132096
    _Float16* lkv  = (_Float16*)(smem + 16 * SLS * 4);           // 18432 (K:128x72 | Vt:64x136)
    _Float16* lq   = (_Float16*)(smem + 16 * SLS * 4 + 18432);   // 16*72*2 = 2304

    const int nh = blockIdx.y;
    const int n = nh >> 4, h = nh & 15;
    const int qbase = blockIdx.x * 16;
    const int tid = threadIdx.x;
    const int wave = tid >> 5, lane = tid & 31;
    const int g = lane >> 4, ln = lane & 15;

    const int kl = qbase + 16;              // valid key columns (causal)
    const int nchunks = (kl + 127) >> 7;    // 128-key chunks
    const int nk = nchunks << 7;

    const size_t headQK = (size_t)nh * SS * DH;

    // ---- stage Q strip 16x64 f16 (rotated, head-major: contiguous rows) ----
    {
        int row = tid >> 4, c2 = (tid & 15) * 4;   // 4 halves per thread
        v2h a = *reinterpret_cast<const v2h*>(qh + headQK + (size_t)(qbase + row) * DH + c2);
        v2h b = *reinterpret_cast<const v2h*>(qh + headQK + (size_t)(qbase + row) * DH + c2 + 2);
        _Float16* d = lq + row * 72 + c2;
        d[0] = a[0]; d[1] = a[1]; d[2] = b[0]; d[3] = b[1];
    }
    __syncthreads();
    const v16h aq0 = load_a_frag(lq, 72, ln, g, 0);
    const v16h aq1 = load_a_frag(lq, 72, ln, g, 32);
    const unsigned lkv_off = (unsigned)(size_t)(void*)lkv;

    // ---- phase 1: scores -> sLog ------------------------------------------
    const float scale = 0.125f;             // 1/sqrt(64)
    for (int c = 0; c < nchunks; ++c) {
        const int kb = c << 7;
        __syncthreads();                    // previous chunk's lkv reads done
        if (tid < 32) {                     // wave 0 drives the TDM
            tdm_load_k_tile(lkv_off, kh + headQK + (size_t)kb * DH, 128);
            __builtin_amdgcn_s_wait_tensorcnt(0);
        }
        __syncthreads();

        v8f acc = {};
        v16h b0 = load_b_frag(lkv, 72, wave * 16 + ln, g, 0);
        v16h b1 = load_b_frag(lkv, 72, wave * 16 + ln, g, 32);
        acc = __builtin_amdgcn_wmma_f32_16x16x32_f16(false, aq0, false, b0,
                                                     (short)0, acc, false, false);
        acc = __builtin_amdgcn_wmma_f32_16x16x32_f16(false, aq1, false, b1,
                                                     (short)0, acc, false, false);
        const int kg = kb + wave * 16 + ln;
#pragma unroll
        for (int r = 0; r < 8; ++r) {
            int qg = qbase + r + 8 * g;
            sLog[(r + 8 * g) * SLS + kg] = (kg <= qg) ? acc[r] * scale : -1e30f;
        }
    }
    __syncthreads();

    // ---- phase 2: softmax over strip rows (2 rows per wave) ---------------
#pragma unroll
    for (int rr = 0; rr < 2; ++rr) {
        const int row = wave * 2 + rr;
        float* lrow = sLog + row * SLS;
        float m = -3.0e38f;
        for (int j = lane; j < nk; j += 32) m = fmaxf(m, lrow[j]);
#pragma unroll
        for (int off = 16; off > 0; off >>= 1) m = fmaxf(m, __shfl_xor(m, off, 32));
        float s = 0.0f;
        for (int j = lane; j < nk; j += 32) {
            float e = __expf(lrow[j] - m);
            lrow[j] = e;
            s += e;
        }
#pragma unroll
        for (int off = 16; off > 0; off >>= 1) s += __shfl_xor(s, off, 32);
        const float inv = 1.0f / s;
        for (int j = lane; j < nk; j += 32) lrow[j] *= inv;
    }
    __syncthreads();

    // ---- phase 3: single write of attn probabilities ----------------------
    float* arow = attn + (size_t)nh * SS * SS + (size_t)qbase * SS;
#pragma unroll
    for (int e = 0; e < 32; ++e) {
        int l4 = tid + (e << 8);            // 8192 float4 = 16x2048
        int row = l4 >> 9;
        int cc = (l4 & 511) << 2;
        float4 o;
        if (cc < nk) {
            const float* p = sLog + row * SLS + cc;
            o = make_float4(p[0], p[1], p[2], p[3]);
        } else {
            o = make_float4(0.f, 0.f, 0.f, 0.f);
        }
        *reinterpret_cast<float4*>(arow + (size_t)row * SS + cc) = o;
    }

    // ---- phase 4: ctx = P @ V, split-K across waves -----------------------
    const int nt = wave & 3, khalf = wave >> 2;   // 4 n-tiles x 2 K-halves
    v8f cacc = {};
    for (int c = 0; c < nchunks; ++c) {
        const int kb = c << 7;
        __syncthreads();                    // previous lVt reads done
#pragma unroll
        for (int e = 0; e < 16; ++e) {      // stage V chunk 128x64, transposed
            int l = tid + (e << 8);
            int krow = l >> 5;
            int c2 = (l & 31) << 1;
            v2h pv = *reinterpret_cast<const v2h*>(
                vh + headQK + (size_t)(kb + krow) * DH + c2);
            lkv[(c2 + 0) * 136 + krow] = pv[0];
            lkv[(c2 + 1) * 136 + krow] = pv[1];
        }
        __syncthreads();
#pragma unroll
        for (int s2 = 0; s2 < 2; ++s2) {
            const int dbase = khalf * 64 + s2 * 32;
            v16h a = a_frag_from_probs(sLog, SLS, ln, g, kb + dbase);
            v16h b = load_b_frag(lkv, 136, nt * 16 + ln, g, dbase);
            cacc = __builtin_amdgcn_wmma_f32_16x16x32_f16(false, a, false, b,
                                                          (short)0, cacc, false, false);
        }
    }
    __syncthreads();
    float* rbuf = sLog;                     // strip no longer needed: reuse
    if (khalf == 1) {
#pragma unroll
        for (int r = 0; r < 8; ++r) rbuf[nt * 256 + r * 32 + lane] = cacc[r];
    }
    __syncthreads();
    if (khalf == 0) {
#pragma unroll
        for (int r = 0; r < 8; ++r) {
            float vsum = cacc[r] + rbuf[nt * 256 + r * 32 + lane];
            ctx[(size_t)(n * SS + qbase + r + 8 * g) * DM + h * DH + nt * 16 + ln] = vsum;
        }
    }
}

// ---------------------------------------------------------------------------
extern "C" void kernel_launch(void* const* d_in, const int* in_sizes, int n_in,
                              void* d_out, int out_size, void* d_ws, size_t ws_size,
                              hipStream_t stream)
{
    const float* x    = (const float*)d_in[0];
    const float* Wq   = (const float*)d_in[1];
    const float* bq   = (const float*)d_in[2];
    const float* Wk   = (const float*)d_in[3];
    const float* bk   = (const float*)d_in[4];
    const float* Wv   = (const float*)d_in[5];
    const float* bv   = (const float*)d_in[6];
    const float* Wo   = (const float*)d_in[7];
    const float* bo   = (const float*)d_in[8];
    const float* cosb = (const float*)d_in[9];
    const float* sinb = (const float*)d_in[10];
    // d_in[11] (mask) is implicit in the causal kernels.

    float* hidden = (float*)d_out;
    float* attn   = hidden + (size_t)NB * SS * DM;

    float*    qf  = (float*)d_ws;
    float*    kf  = qf  + (size_t)MT * DM;
    float*    vf  = kf  + (size_t)MT * DM;
    float*    ctx = vf  + (size_t)MT * DM;
    _Float16* qh  = (_Float16*)(ctx + (size_t)MT * DM);
    _Float16* kh  = qh + (size_t)MT * DM;
    _Float16* vh  = kh + (size_t)MT * DM;

    dim3 gGemm(DM / 64, MT / 64);          // (16, 64)
    gemm_bias_kernel<<<gGemm, 256, 0, stream>>>(x, Wq, bq, qf, MT, DM, DM);
    gemm_bias_kernel<<<gGemm, 256, 0, stream>>>(x, Wk, bk, kf, MT, DM, DM);
    gemm_bias_kernel<<<gGemm, 256, 0, stream>>>(x, Wv, bv, vf, MT, DM, DM);

    int ropeTotal = NB * SS * NH * (DH / 2);   // 2,097,152
    rope_pack_kernel<<<ropeTotal / 256, 256, 0, stream>>>(qf, qh, cosb, sinb);
    rope_pack_kernel<<<ropeTotal / 256, 256, 0, stream>>>(kf, kh, cosb, sinb);
    pack_v_kernel<<<ropeTotal / 256, 256, 0, stream>>>(vf, vh);

    size_t smem = 16 * SLS * 4 + 18432 + 2304;   // 152832 B (<320KB WGP LDS)
    attn_fused_kernel<<<dim3(SS / 16, NB * NH), 256, smem, stream>>>(
        qh, kh, vh, attn, ctx);

    gemm_bias_kernel<<<gGemm, 256, 0, stream>>>(ctx, Wo, bo, hidden, MT, DM, DM);
}
